// BVRNN_6055903887535
// MI455X (gfx1250) — matrix operand order repaired and checked by reading, not compiled, not run on hardware
//
#include <hip/hip_runtime.h>

// ---------------------------------------------------------------------------
// BVRNN forward for MI455X (gfx1250), wave32 + V_WMMA_F32_16X16X32_F16.
// Phase 1: phi_x MLP over all (B,T) rows as big WMMA GEMMs (NW=4 strips).
// Phase 2: persistent-kernel scan over T=600 with global software barriers.
// Compile-time K shapes -> fully unrolled; global-AS loads (global_load_b128);
// explicit double-buffered fragment pipeline so each WMMA waits only on its
// own operands while next k-step loads are in flight; 32-bit address math.
// ---------------------------------------------------------------------------

#define ACT_NONE 0
#define ACT_ELU  1
#define ACT_SIG  2

typedef __attribute__((ext_vector_type(16))) _Float16 v16h;
typedef __attribute__((ext_vector_type(8)))  _Float16 v8h;
typedef __attribute__((ext_vector_type(8)))  float    v8f;

#define GAS __attribute__((address_space(1)))

#define BB   64
#define TT   600
#define XD   80
#define XP   96          // X padded to multiple of 32 for WMMA K
#define HD   512
#define ZD   64
#define GDIM 1536
#define ROWS1 (BB*TT)    // 38400

// ---- workspace layout: half region, then float region -----------------------
enum HOFF : long long {
  O_WPX0 = 0,                       // [512][96]
  O_WPX1 = O_WPX0 + 512*96,         // [512][512]
  O_WPX2 = O_WPX1 + 512*512,
  O_WPZ0 = O_WPX2 + 512*512,        // [512][64]
  O_WPZ1 = O_WPZ0 + 512*64,
  O_WPZ2 = O_WPZ1 + 512*512,
  O_WE0  = O_WPZ2 + 512*512,        // [512][1024]
  O_WE1  = O_WE0  + 512*1024,
  O_WE2  = O_WE1  + 512*512,        // [64][512]
  O_WP0  = O_WE2  + 64*512,
  O_WP1  = O_WP0  + 512*512,
  O_WP2  = O_WP1  + 512*512,        // [64][512]
  O_WD0  = O_WP2  + 64*512,         // [512][1024]
  O_WD1  = O_WD0  + 512*1024,
  O_WD2  = O_WD1  + 512*512,
  O_WD3  = O_WD2  + 512*512,        // [80][512]
  O_WGI  = O_WD3  + 80*512,         // [1536][1024]
  O_WGH  = O_WGI  + 1536*1024,      // [1536][512]
  O_H16  = O_WGH  + 1536*512,       // h as f16 [64][512]
  O_SA   = O_H16  + 64*512,         // generic f16 act buffers [64][512]
  O_SB   = O_SA   + 64*512,
  O_PA   = O_SB   + 64*512,
  O_PB   = O_PA   + 64*512,
  O_Z16  = O_PB   + 64*512,         // round(enc) [64][64]
  O_DN   = O_Z16  + 64*64,          // normalized dec [64][96] (pad zeroed)
  O_YN   = O_DN   + 64*96,          // normalized y [38400][96]
  O_P1A  = O_YN   + (long long)ROWS1*96,
  O_P1B  = O_P1A  + (long long)ROWS1*512,
  O_PHIX = O_P1B  + (long long)ROWS1*512,   // phi_x as [t][b][512]
  H_END  = O_PHIX + (long long)ROWS1*512,
};

enum FOFF : long long {
  F_H32  = 0,                 // h f32 [64][512]
  F_ENC  = F_H32 + 64*512,    // enc_t [64][64]
  F_PRI  = F_ENC + 64*64,     // prior_t [64][64]
  F_GI   = F_PRI + 64*64,     // gi [64][1536]
  F_GH   = F_GI  + 64*1536,   // gh [64][1536]
  F_SYNC = F_GH  + 64*1536,   // 2 ints: barrier counter + sense
  F_END  = F_SYNC + 2,
};

static constexpr size_t HALF_BYTES = (((size_t)H_END * 2) + 255) & ~((size_t)255);

// ---- generic -> global address-space casts (buffers are all global memory) --
__device__ inline const GAS _Float16* gch(const _Float16* p) {
  return (const GAS _Float16*)(unsigned long long)p;
}
__device__ inline GAS _Float16* gh_(_Float16* p) {
  return (GAS _Float16*)(unsigned long long)p;
}
__device__ inline const GAS float* gcf(const float* p) {
  return (const GAS float*)(unsigned long long)p;
}
__device__ inline GAS float* gf_(float* p) {
  return (GAS float*)(unsigned long long)p;
}

// ---------------------------------------------------------------------------
struct GemmD {
  const _Float16* A0;   // [M][lda0]
  const _Float16* A1;   // optional second K-segment (virtual concat)
  const _Float16* Bt;   // [N][ldb]   weight, transposed, K contiguous
  const float*    bias; // [N]
  float*          o32;  // optional f32 out
  _Float16*       o16;  // optional f16 out
  const float*    nm;   // normalize mean (flags&2)
  const float*    ns;   // normalize std  (flags&2)
  int lda0, lda1, ldb;
  int N;
  int ldo32, ldo16;
  int act, flags;       // flags: 1=rint before f16, 2=normalize, 4=scatter [t][b]
};

__host__ __device__ inline GemmD mkgd(const _Float16* A0, int lda0,
                                      const _Float16* A1, int lda1,
                                      const _Float16* Bt, int ldb,
                                      const float* bias, int N, int act,
                                      float* o32, int ldo32,
                                      _Float16* o16, int ldo16, int flags,
                                      const float* nm, const float* ns) {
  GemmD g;
  g.A0 = A0; g.A1 = A1; g.Bt = Bt; g.bias = bias;
  g.o32 = o32; g.o16 = o16; g.nm = nm; g.ns = ns;
  g.lda0 = lda0; g.lda1 = lda1; g.ldb = ldb; g.N = N;
  g.ldo32 = ldo32; g.ldo16 = ldo16; g.act = act; g.flags = flags;
  return g;
}

__device__ inline float elu1(float v) { return v > 0.f ? v : (__expf(v) - 1.f); }
__device__ inline float sigm(float v) { return 1.f / (1.f + __expf(-v)); }

// A 16x32 f16 fragment: lanes 0-15 hold K {0..7,16..23}, lanes 16-31 {8..15,24..31}
__device__ inline v16h load_frag_a(const GAS _Float16* p) {
  const GAS v8h* q = (const GAS v8h*)p;
  v8h lo = q[0];
  v8h hi = q[2];   // +16 halves
  return __builtin_shufflevector(lo, hi, 0,1,2,3,4,5,6,7,8,9,10,11,12,13,14,15);
}
// B 32x16 f16 fragment: lanes 0-15 hold K {0..7,16..23}, lanes 16-31 {8..15,24..31} of col (lane&15)
__device__ inline v16h load_frag_b(const GAS _Float16* p) {
  const GAS v8h* q = (const GAS v8h*)p;
  v8h lo = q[0];
  v8h hi = q[2];   // +16 halves
  return __builtin_shufflevector(lo, hi, 0,1,2,3,4,5,6,7,8,9,10,11,12,13,14,15);
}

// One strip of NW adjacent 16x16 output tiles. K0E/K1E: compile-time K segment
// lengths (K1E=0 -> no concat). Fully unrolled with an explicit 1-deep
// software pipeline: fragments for step kt+1 are issued before the WMMAs of
// step kt, so each WMMA waits only on its own operands.
template<int K0E, int K1E, int NW>
__device__ void tile_strip(const GemmD& g, int strip) {
  constexpr int NKT = (K0E + K1E) / 32;
  const int nsn = g.N / (16 * NW);
  const int mt = strip / nsn;
  const int ns = strip - mt * nsn;
  const int lane = threadIdx.x & 31;
  const int hi   = lane >> 4;
  const int l15  = lane & 15;
  const int arow = mt * 16 + l15;
  const int koff = hi ? 8 : 0;

  // 32-bit offset math (all buffers < 2^31 elements) -> v_mul_lo_u32 not u64
  const GAS _Float16* Ap0 = gch(g.A0) + (unsigned)(arow * g.lda0 + koff);
  const GAS _Float16* Ap1 =
      (K1E > 0) ? (gch(g.A1) + (unsigned)(arow * g.lda1 + koff))
                : (const GAS _Float16*)0;
  const GAS _Float16* Bp[NW];
#pragma unroll
  for (int w = 0; w < NW; ++w)
    Bp[w] = gch(g.Bt) + (unsigned)(((ns * NW + w) * 16 + l15) * g.ldb + (hi ? 8 : 0));

  // kt is a compile-time constant after unrolling: segment select folds away.
  auto loadA = [&](int kt) -> v16h {
    return (kt * 32 < K0E) ? load_frag_a(Ap0 + kt * 32)
                           : load_frag_a(Ap1 + (kt * 32 - K0E));
  };

  v8f acc[NW];
#pragma unroll
  for (int w = 0; w < NW; ++w) acc[w] = v8f{0.f,0.f,0.f,0.f,0.f,0.f,0.f,0.f};

  v16h a_cur = loadA(0);
  v16h b_cur[NW];
#pragma unroll
  for (int w = 0; w < NW; ++w) b_cur[w] = load_frag_b(Bp[w]);

#pragma unroll
  for (int kt = 0; kt < NKT; ++kt) {
    v16h a_nxt = a_cur;
    v16h b_nxt[NW];
#pragma unroll
    for (int w = 0; w < NW; ++w) b_nxt[w] = b_cur[w];
    if (kt + 1 < NKT) {
      a_nxt = loadA(kt + 1);
#pragma unroll
      for (int w = 0; w < NW; ++w) b_nxt[w] = load_frag_b(Bp[w] + (kt + 1) * 32);
    }
#pragma unroll
    for (int w = 0; w < NW; ++w)
      acc[w] = __builtin_amdgcn_wmma_f32_16x16x32_f16(false, a_cur, false, b_cur[w],
                                                      (short)0, acc[w], false, false);
    a_cur = a_nxt;
#pragma unroll
    for (int w = 0; w < NW; ++w) b_cur[w] = b_nxt[w];
  }

  // C layout: VGPR r, lane l -> N = l&15, M = r + (l>=16 ? 8 : 0)
  GAS float*    o32 = g.o32 ? gf_(g.o32) : (GAS float*)0;
  GAS _Float16* o16 = g.o16 ? gh_(g.o16) : (GAS _Float16*)0;
#pragma unroll
  for (int w = 0; w < NW; ++w) {
    const int n = (ns * NW + w) * 16 + l15;
    const float bv = g.bias ? gcf(g.bias)[n] : 0.f;
#pragma unroll
    for (int r = 0; r < 8; ++r) {
      const int m = mt * 16 + r + (hi ? 8 : 0);
      float v = acc[w][r] + bv;
      if (g.act == ACT_ELU)      v = elu1(v);
      else if (g.act == ACT_SIG) v = sigm(v);
      if (o32) o32[(size_t)m * g.ldo32 + n] = v;
      if (o16) {
        float h = v;
        if (g.flags & 1) h = rintf(h);                       // straight-through round
        if (g.flags & 2) h = (h - gcf(g.nm)[n]) / gcf(g.ns)[n];  // dec normalize
        size_t idx;
        if (g.flags & 4) {                                   // phase1 scatter [t][b][h]
          const int bbi = m / TT, tti = m - bbi * TT;
          idx = ((size_t)tti * BB + bbi) * g.ldo16 + n;
        } else {
          idx = (size_t)m * g.ldo16 + n;
        }
        o16[idx] = (_Float16)h;
      }
    }
  }
}

// ---------------------------------------------------------------------------
template<int K0E, int K1E, int NW>
__global__ __launch_bounds__(256) void gemm_kernel(GemmD g, int nstrips) {
  const int nw = gridDim.x * (blockDim.x >> 5);
  const int gw = blockIdx.x * (blockDim.x >> 5) + (threadIdx.x >> 5);
  for (int i = gw; i < nstrips; i += nw) tile_strip<K0E, K1E, NW>(g, i);
}

__global__ void convert_wt(const float* __restrict__ W, _Float16* __restrict__ out,
                           int K, int N, int Kp) {
  int i = blockIdx.x * blockDim.x + threadIdx.x;
  if (i >= N * Kp) return;
  int n = i / Kp, k = i - n * Kp;
  gh_(out)[i] = (k < K) ? (_Float16)gcf(W)[(size_t)k * N + n] : (_Float16)0.f;
}

__global__ void pack_yn(const float* __restrict__ y, const float* __restrict__ mean,
                        const float* __restrict__ stdv, _Float16* __restrict__ out) {
  int i = blockIdx.x * blockDim.x + threadIdx.x;
  if (i >= ROWS1 * XP) return;
  int r = i / XP, c = i - r * XP;
  gh_(out)[i] = (c < XD)
      ? (_Float16)((gcf(y)[(size_t)r * XD + c] - gcf(mean)[c]) / gcf(stdv)[c])
      : (_Float16)0.f;
}

__global__ void init_kernel(char* ws) {
  _Float16* hb = (_Float16*)ws;
  float* fb = (float*)(ws + HALF_BYTES);
  int i = blockIdx.x * blockDim.x + threadIdx.x;
  if (i < BB * HD) { gf_(fb)[F_H32 + i] = 0.f; gh_(hb)[O_H16 + i] = (_Float16)0.f; }
  if (i < BB * XP) gh_(hb)[O_DN + i] = (_Float16)0.f;   // keeps K-pad columns zero
  if (i == 0) { int* sy = (int*)(fb + F_SYNC); sy[0] = 0; sy[1] = 0; }
}

// Sense-reversing device-wide barrier (all WGs resident: persistent kernel).
__device__ void gbar(int* sy, int nwg) {
  __syncthreads();
  if (threadIdx.x == 0) {
    __threadfence();
    int gen = __hip_atomic_load(&sy[1], __ATOMIC_RELAXED, __HIP_MEMORY_SCOPE_AGENT);
    int v = __hip_atomic_fetch_add(&sy[0], 1, __ATOMIC_ACQ_REL, __HIP_MEMORY_SCOPE_AGENT);
    if (v == nwg - 1) {
      __hip_atomic_store(&sy[0], 0, __ATOMIC_RELAXED, __HIP_MEMORY_SCOPE_AGENT);
      __hip_atomic_fetch_add(&sy[1], 1, __ATOMIC_RELEASE, __HIP_MEMORY_SCOPE_AGENT);
    } else {
      while (__hip_atomic_load(&sy[1], __ATOMIC_ACQUIRE, __HIP_MEMORY_SCOPE_AGENT) == gen) {
        __builtin_amdgcn_s_sleep(8);
      }
    }
    __threadfence();
  }
  __syncthreads();
}

struct ScanP {
  char* ws; float* out;
  const float* mean; const float* stdv;
  const float *be0, *be1, *be2;       // enc biases
  const float *bp0, *bp1, *bp2;       // prior biases
  const float *bz0, *bz1, *bz2;       // phi_z biases
  const float *bx0, *bx1, *bx2;       // phi_x biases
  const float *bd0, *bd1, *bd2, *bd3; // dec biases
  const float *bih, *bhh;             // gru biases
};

template<int K0E, int K1E, int NW>
__device__ inline void run1(const GemmD& g, int nstrips, int gwave, int nwaves) {
  for (int i = gwave; i < nstrips; i += nwaves) tile_strip<K0E, K1E, NW>(g, i);
}
template<int K0a, int K1a, int NWa, int K0b, int K1b, int NWb>
__device__ inline void run2(const GemmD& a, int na, const GemmD& b, int nb,
                            int gwave, int nwaves) {
  const int tot = na + nb;
  for (int i = gwave; i < tot; i += nwaves) {
    if (i < na) tile_strip<K0a, K1a, NWa>(a, i);
    else        tile_strip<K0b, K1b, NWb>(b, i - na);
  }
}

#define NT512 128   // (64/16)*(512/16), NW=1
#define NT64  16
#define NT80  20
#define NTG3  128   // (64/16)*(1536/48), NW=3

__global__ __launch_bounds__(256) void scan_kernel(ScanP p) {
  _Float16* hb = (_Float16*)p.ws;
  float* fb = (float*)(p.ws + HALF_BYTES);
  int* sy = (int*)(fb + F_SYNC);
  const int nwg = gridDim.x;
  const int nwaves = gridDim.x * (blockDim.x >> 5);
  const int gwave  = blockIdx.x * (blockDim.x >> 5) + (threadIdx.x >> 5);
  const int nthr = gridDim.x * blockDim.x;
  const int gtid = blockIdx.x * blockDim.x + threadIdx.x;
  __shared__ float red[256];

  _Float16* H16 = hb + O_H16;
  _Float16* SA  = hb + O_SA;  _Float16* SB = hb + O_SB;
  _Float16* PA  = hb + O_PA;  _Float16* PB = hb + O_PB;
  _Float16* Z16 = hb + O_Z16; _Float16* DN = hb + O_DN;
  const _Float16* WPX0 = hb + O_WPX0; const _Float16* WPX1 = hb + O_WPX1;
  const _Float16* WPX2 = hb + O_WPX2;
  const _Float16* WPZ0 = hb + O_WPZ0; const _Float16* WPZ1 = hb + O_WPZ1;
  const _Float16* WPZ2 = hb + O_WPZ2;
  const _Float16* WE0 = hb + O_WE0; const _Float16* WE1 = hb + O_WE1;
  const _Float16* WE2 = hb + O_WE2;
  const _Float16* WP0 = hb + O_WP0; const _Float16* WP1 = hb + O_WP1;
  const _Float16* WP2 = hb + O_WP2;
  const _Float16* WD0 = hb + O_WD0; const _Float16* WD1 = hb + O_WD1;
  const _Float16* WD2 = hb + O_WD2; const _Float16* WD3 = hb + O_WD3;
  const _Float16* WGI = hb + O_WGI; const _Float16* WGH = hb + O_WGH;
  float* H32 = fb + F_H32; float* E32 = fb + F_ENC; float* P32 = fb + F_PRI;
  float* GI = fb + F_GI;  float* GH = fb + F_GH;

  float kacc = 0.f;  // KLD accumulator (meaningful only on block0/thread0)

  for (int t = 0; t < TT; ++t) {
    const _Float16* PHX = hb + O_PHIX + (size_t)t * (BB * HD);

    // S1: enc0 = elu([phi_x_t, h] @ We0)  ||  prior0 = elu(h @ Wp0)
    run2<512,512,1, 512,0,1>(
        mkgd(PHX,HD, H16,HD, WE0,2*HD, p.be0, HD, ACT_ELU,
             nullptr,0, SA,HD, 0, nullptr,nullptr), NT512,
        mkgd(H16,HD, nullptr,0, WP0,HD, p.bp0, HD, ACT_ELU,
             nullptr,0, PA,HD, 0, nullptr,nullptr), NT512, gwave, nwaves);
    gbar(sy, nwg);
    // S2
    run2<512,0,1, 512,0,1>(
        mkgd(SA,HD, nullptr,0, WE1,HD, p.be1, HD, ACT_ELU,
             nullptr,0, SB,HD, 0, nullptr,nullptr), NT512,
        mkgd(PA,HD, nullptr,0, WP1,HD, p.bp1, HD, ACT_ELU,
             nullptr,0, PB,HD, 0, nullptr,nullptr), NT512, gwave, nwaves);
    gbar(sy, nwg);
    // S3: enc_t = sigmoid -> f32 + z = rint(enc_t) f16 || prior_t -> f32
    run2<512,0,1, 512,0,1>(
        mkgd(SB,HD, nullptr,0, WE2,HD, p.be2, ZD, ACT_SIG,
             E32,ZD, Z16,ZD, 1, nullptr,nullptr), NT64,
        mkgd(PB,HD, nullptr,0, WP2,HD, p.bp2, ZD, ACT_SIG,
             P32,ZD, nullptr,0, 0, nullptr,nullptr), NT64, gwave, nwaves);
    gbar(sy, nwg);
    // S4-S6: phi_z chain (z -> SA -> SB -> PA = phi_z_t)
    run1<64,0,1>(mkgd(Z16,ZD, nullptr,0, WPZ0,ZD, p.bz0, HD, ACT_ELU,
                      nullptr,0, SA,HD, 0, nullptr,nullptr), NT512, gwave, nwaves);
    gbar(sy, nwg);
    run1<512,0,1>(mkgd(SA,HD, nullptr,0, WPZ1,HD, p.bz1, HD, ACT_ELU,
                       nullptr,0, SB,HD, 0, nullptr,nullptr), NT512, gwave, nwaves);
    gbar(sy, nwg);
    run1<512,0,1>(mkgd(SB,HD, nullptr,0, WPZ2,HD, p.bz2, HD, ACT_ELU,
                       nullptr,0, PA,HD, 0, nullptr,nullptr), NT512, gwave, nwaves);
    gbar(sy, nwg);
    // S7-S9: dec chain ([phi_z_t, h] -> SA -> SB -> SA)
    run1<512,512,1>(mkgd(PA,HD, H16,HD, WD0,2*HD, p.bd0, HD, ACT_ELU,
                         nullptr,0, SA,HD, 0, nullptr,nullptr), NT512, gwave, nwaves);
    gbar(sy, nwg);
    run1<512,0,1>(mkgd(SA,HD, nullptr,0, WD1,HD, p.bd1, HD, ACT_ELU,
                       nullptr,0, SB,HD, 0, nullptr,nullptr), NT512, gwave, nwaves);
    gbar(sy, nwg);
    run1<512,0,1>(mkgd(SB,HD, nullptr,0, WD2,HD, p.bd2, HD, ACT_ELU,
                       nullptr,0, SA,HD, 0, nullptr,nullptr), NT512, gwave, nwaves);
    gbar(sy, nwg);
    // S10: dec_t -> d_out[b][t][x] (f32) and normalized f16 into DN (K-padded)
    run1<512,0,1>(mkgd(SA,HD, nullptr,0, WD3,HD, p.bd3, XD, ACT_NONE,
                       p.out + (size_t)t * XD, TT*XD, DN,XP, 2, p.mean, p.stdv),
                  NT80, gwave, nwaves);
    gbar(sy, nwg);
    // S11-S13: phi_x_gen chain (DN -> SB -> SA -> PB = phi_x_gen)
    run1<96,0,1>(mkgd(DN,XP, nullptr,0, WPX0,XP, p.bx0, HD, ACT_ELU,
                      nullptr,0, SB,HD, 0, nullptr,nullptr), NT512, gwave, nwaves);
    gbar(sy, nwg);
    run1<512,0,1>(mkgd(SB,HD, nullptr,0, WPX1,HD, p.bx1, HD, ACT_ELU,
                       nullptr,0, SA,HD, 0, nullptr,nullptr), NT512, gwave, nwaves);
    gbar(sy, nwg);
    run1<512,0,1>(mkgd(SA,HD, nullptr,0, WPX2,HD, p.bx2, HD, ACT_ELU,
                       nullptr,0, PB,HD, 0, nullptr,nullptr), NT512, gwave, nwaves);
    gbar(sy, nwg);
    // S14: gi = [phi_x_gen, phi_z_t] @ w_ih  ||  gh = h @ w_hh   (NW=3 strips,
    // 128+128 strips == one strip per wave across the whole grid)
    run2<512,512,3, 512,0,3>(
        mkgd(PB,HD, PA,HD, WGI,2*HD, p.bih, GDIM, ACT_NONE,
             GI,GDIM, nullptr,0, 0, nullptr,nullptr), NTG3,
        mkgd(H16,HD, nullptr,0, WGH,HD, p.bhh, GDIM, ACT_NONE,
             GH,GDIM, nullptr,0, 0, nullptr,nullptr), NTG3, gwave, nwaves);
    gbar(sy, nwg);

    // S15: GRU gates + h update (all blocks), KLD reduction (block 0)
    {
      const GAS float* gGI = gcf(GI); const GAS float* gGH = gcf(GH);
      GAS float* gH32 = gf_(H32);     GAS _Float16* gH16 = gh_(H16);
      for (int i = gtid; i < BB * HD; i += nthr) {
        const int bi = i >> 9, j = i & (HD - 1);
        const GAS float* gi = gGI + (size_t)bi * GDIM;
        const GAS float* gh = gGH + (size_t)bi * GDIM;
        const float r  = sigm(gi[j] + gh[j]);
        const float z  = sigm(gi[HD + j] + gh[HD + j]);
        const float nn = tanhf(gi[2*HD + j] + r * gh[2*HD + j]);
        const float hv = gH32[i];
        const float hnew = (1.f - z) * nn + z * hv;
        gH32[i] = hnew;
        gH16[i] = (_Float16)hnew;
      }
    }
    if (blockIdx.x == 0) {
      const GAS float* gE = gcf(E32); const GAS float* gP = gcf(P32);
      float s = 0.f;
      for (int i = threadIdx.x; i < BB * ZD; i += blockDim.x) {
        const float e = gE[i], q = gP[i], c = 0.001f;
        s += e * (__logf(fmaxf(e, c)) - __logf(fmaxf(q, c)))
           + (1.f - e) * (__logf(fmaxf(1.f - e, c)) - __logf(fmaxf(1.f - q, c)));
      }
      red[threadIdx.x] = s; __syncthreads();
      for (int st = 128; st > 0; st >>= 1) {
        if ((int)threadIdx.x < st) red[threadIdx.x] += red[threadIdx.x + st];
        __syncthreads();
      }
      if (threadIdx.x == 0) kacc += red[0] * (1.f / BB);
    }
    gbar(sy, nwg);
  }

  if (blockIdx.x == 0 && threadIdx.x == 0)
    gf_(p.out)[(size_t)BB * TT * XD] = kacc * (1.f / TT);
}

// ---------------------------------------------------------------------------
extern "C" void kernel_launch(void* const* d_in, const int* in_sizes, int n_in,
                              void* d_out, int out_size, void* d_ws, size_t ws_size,
                              hipStream_t stream) {
  (void)in_sizes; (void)n_in; (void)out_size; (void)ws_size;
  char* ws = (char*)d_ws;
  _Float16* hb = (_Float16*)ws;
  auto F = [&](int i) { return (const float*)d_in[i]; };
  const float* y = F(0); const float* mean = F(1); const float* stdv = F(2);

  init_kernel<<<(BB*HD + 255) / 256, 256, 0, stream>>>(ws);

  struct C { int idx; int K, N, Kp; long long off; };
  const C cv[] = {
    {3, 80, 512, 96, O_WPX0}, {5, 512, 512, 512, O_WPX1}, {7, 512, 512, 512, O_WPX2},
    {9, 64, 512, 64, O_WPZ0}, {11,512, 512, 512, O_WPZ1}, {13,512, 512, 512, O_WPZ2},
    {15,1024,512,1024,O_WE0}, {17,512, 512, 512, O_WE1},  {19,512, 64, 512, O_WE2},
    {21,512, 512, 512, O_WP0},{23,512, 512, 512, O_WP1},  {25,512, 64, 512, O_WP2},
    {27,1024,512,1024,O_WD0}, {29,512, 512, 512, O_WD1},  {31,512, 512, 512, O_WD2},
    {33,512, 80, 512, O_WD3}, {35,1024,1536,1024,O_WGI},  {36,512, 1536,512, O_WGH},
  };
  for (const C& c : cv) {
    const int tot = c.N * c.Kp;
    convert_wt<<<(tot + 255) / 256, 256, 0, stream>>>(F(c.idx), hb + c.off,
                                                      c.K, c.N, c.Kp);
  }

  pack_yn<<<(ROWS1 * XP + 255) / 256, 256, 0, stream>>>(y, mean, stdv, hb + O_YN);

  // Phase 1: phi_x over all 38400 rows (3 GEMMs, NW=4 strips, full occupancy)
  const int P1S = (ROWS1 / 16) * (HD / 64);   // 2400 * 8 = 19200 strips
  GemmD g1 = mkgd(hb + O_YN, XP, nullptr, 0, hb + O_WPX0, XP, F(4), HD,
                  ACT_ELU, nullptr, 0, hb + O_P1A, HD, 0, nullptr, nullptr);
  gemm_kernel<96, 0, 4><<<1200, 256, 0, stream>>>(g1, P1S);
  GemmD g2 = mkgd(hb + O_P1A, HD, nullptr, 0, hb + O_WPX1, HD, F(6), HD,
                  ACT_ELU, nullptr, 0, hb + O_P1B, HD, 0, nullptr, nullptr);
  gemm_kernel<512, 0, 4><<<1200, 256, 0, stream>>>(g2, P1S);
  GemmD g3 = mkgd(hb + O_P1B, HD, nullptr, 0, hb + O_WPX2, HD, F(8), HD,
                  ACT_ELU, nullptr, 0, hb + O_PHIX, HD, 4 /*scatter [t][b]*/,
                  nullptr, nullptr);
  gemm_kernel<512, 0, 4><<<1200, 256, 0, stream>>>(g3, P1S);

  // Phase 2: persistent scan kernel
  ScanP sp;
  sp.ws = ws; sp.out = (float*)d_out; sp.mean = mean; sp.stdv = stdv;
  sp.be0 = F(16); sp.be1 = F(18); sp.be2 = F(20);
  sp.bp0 = F(22); sp.bp1 = F(24); sp.bp2 = F(26);
  sp.bz0 = F(10); sp.bz1 = F(12); sp.bz2 = F(14);
  sp.bx0 = F(4);  sp.bx1 = F(6);  sp.bx2 = F(8);
  sp.bd0 = F(28); sp.bd1 = F(30); sp.bd2 = F(32); sp.bd3 = F(34);
  sp.bih = F(37); sp.bhh = F(38);
  scan_kernel<<<32, 256, 0, stream>>>(sp);
}
